// Conv2D_26164940767465
// MI455X (gfx1250) — 8-bit matrix operand order repaired and checked by reading, compiled, not run on hardware
//
#include <hip/hip_runtime.h>

typedef __attribute__((ext_vector_type(8))) int v8i;

#define BATCH 4
#define H     224
#define W     224
#define CIN   16
#define FOUT  32
#define KH    5
#define KW    5
#define HOUT  220
#define WOUT  220
#define ROWB  (KH * W)

__global__ void pack_weights_kernel(const float* __restrict__ kern,
                                    int* __restrict__ wsA) {
    int id = blockIdx.x * blockDim.x + threadIdx.x;
    if (id >= CIN * 2 * 32) return;
    const int c    = id >> 6;
    const int t    = (id >> 5) & 1;
    const int L    = id & 31;
    const int m    = L & 15;
    const int half = L >> 4;
    const int f    = t * 16 + m;
#pragma unroll
    for (int v = 0; v < 8; ++v) {
        const int kbase = ((v & 3) << 2) + (half << 4)
                        + ((v >> 2) << 5);
        unsigned int dw = 0;
#pragma unroll
        for (int bb = 0; bb < 4; ++bb) {
            const int kk = kbase + bb;
            if (kk < KH * KW) {
                const int i = kk / KW, j = kk % KW;
                const int kv = (int)kern[((i * KW + j) * CIN + c) * FOUT + f];
                dw |= ((unsigned int)(kv & 0xFF)) << (8 * bb);
            }
        }
        wsA[id * 8 + v] = (int)dw;
    }
}

__device__ __forceinline__ v8i build_b(const unsigned int* __restrict__ xbuf32,
                                       int cbase, int wo0, int n, int half) {
    unsigned int lo[KH], b4[KH];
#pragma unroll
    for (int i = 0; i < KH; ++i) {
        const int addr = cbase + i * W + wo0 + n;
        const int a2   = addr >> 2;
        const int sh   = (addr & 3) * 8;
        const unsigned int d0 = xbuf32[a2];
        const unsigned int d1 = xbuf32[a2 + 1];
        const unsigned long long dd = ((unsigned long long)d1 << 32) | d0;
        lo[i] = (unsigned int)(dd >> sh);
        b4[i] = (d1 >> sh) & 0xFFu;
    }
    const unsigned int s0 = lo[0];
    const unsigned int s1 = b4[0] | (lo[1] << 8);
    const unsigned int s2 = (lo[1] >> 24) | (b4[1] << 8) | ((lo[2] & 0xFFFFu) << 16);
    const unsigned int s3 = ((lo[2] >> 16) & 0xFFFFu) | (b4[2] << 16) | (lo[3] << 24);
    const unsigned int s4 = (lo[3] >> 8) | (b4[3] << 24);
    const unsigned int s5 = lo[4];
    const unsigned int s6 = b4[4];

    v8i bm;
    bm[0] = (int)(half ? s4 : s0);
    bm[1] = (int)(half ? s5 : s1);
    bm[2] = (int)(half ? s6 : s2);
    bm[3] = (int)(half ? 0u : s3);
    bm[4] = 0; bm[5] = 0; bm[6] = 0; bm[7] = 0;
    return bm;
}

__global__ __launch_bounds__(512)
void conv2d_perchan_wmma_kernel(const float* __restrict__ x,
                                const int* __restrict__ wsA,
                                float* __restrict__ out) {
    __shared__ unsigned int xbuf32[(CIN * ROWB + 32) / 4];
    unsigned char* xbuf = (unsigned char*)xbuf32;

    const int bh   = blockIdx.x;
    const int b    = bh / HOUT;
    const int ho   = bh % HOUT;
    const int lane = threadIdx.x;
    const int c    = threadIdx.y;

    const float* xrow = x + (size_t)(b * H + ho) * (W * CIN);
    for (int e = c * 32 + lane; e < KH * W * CIN; e += 512) {
        const int r   = e / (W * CIN);
        const int rem = e - r * (W * CIN);
        const int w   = rem >> 4;
        const int cc  = rem & 15;
        xbuf[cc * ROWB + r * W + w] = (unsigned char)(int)xrow[e];
    }
    __syncthreads();

    v8i a0, a1;
    {
        const int* pa0 = wsA + ((size_t)(c * 2 + 0) * 32 + lane) * 8;
        const int* pa1 = wsA + ((size_t)(c * 2 + 1) * 32 + lane) * 8;
#pragma unroll
        for (int v = 0; v < 8; ++v) { a0[v] = pa0[v]; a1[v] = pa1[v]; }
    }

    const int n     = lane & 15;
    const int half  = lane >> 4;
    const int cbase = c * ROWB;
    float* outc = out + ((size_t)(b * HOUT + ho) * CIN + c) * FOUT * (size_t)WOUT
                + (size_t)(8 * half) * WOUT + n;

    v8i bm = build_b(xbuf32, cbase, 0, n, half);
#pragma unroll 1
    for (int tile = 0; tile < 13; ++tile) {
        v8i zero = {};
        v8i d0 = __builtin_amdgcn_wmma_i32_16x16x64_iu8(true, a0, false, bm, zero, false, false);
        v8i d1 = __builtin_amdgcn_wmma_i32_16x16x64_iu8(true, a1, false, bm, zero, false, false);
        bm = build_b(xbuf32, cbase, (tile + 1) * 16, n, half);
        float* p = outc + tile * 16;
#pragma unroll
        for (int r = 0; r < 8; ++r) {
            p[(size_t)r * WOUT]        = (float)d0[r];
            p[(size_t)(r + 16) * WOUT] = (float)d1[r];
        }
    }
    {
        v8i zero = {};
        v8i d0 = __builtin_amdgcn_wmma_i32_16x16x64_iu8(true, a0, false, bm, zero, false, false);
        v8i d1 = __builtin_amdgcn_wmma_i32_16x16x64_iu8(true, a1, false, bm, zero, false, false);
        if (208 + n < WOUT) {
            float* p = outc + 13 * 16;
#pragma unroll
            for (int r = 0; r < 8; ++r) {
                p[(size_t)r * WOUT]        = (float)d0[r];
                p[(size_t)(r + 16) * WOUT] = (float)d1[r];
            }
        }
    }
}

extern "C" void kernel_launch(void* const* d_in, const int* in_sizes, int n_in,
                              void* d_out, int out_size, void* d_ws, size_t ws_size,
                              hipStream_t stream) {
    const float* x    = (const float*)d_in[0];
    const float* kern = (const float*)d_in[1];
    float* out        = (float*)d_out;
    int* wsA          = (int*)d_ws;

    pack_weights_kernel<<<4, 256, 0, stream>>>(kern, wsA);

    dim3 grid(BATCH * HOUT);
    dim3 block(32, 16);
    conv2d_perchan_wmma_kernel<<<grid, block, 0, stream>>>(x, wsA, out);
}
